// AdvancedGRU_28037546508361
// MI455X (gfx1250) — compile-verified, matrix operand order checked by reading, not run on hardware
//
#include <hip/hip_runtime.h>
#include <hip/hip_bf16.h>
#include <math.h>

// ---------------------------------------------------------------------------
// AdvancedGRU on MI455X (gfx1250): bf16 WMMA recurrent steps, f32 state,
// async global->LDS staging of the activation panels (CDNA5 ASYNCcnt path).
//
// Shapes: B=64, S=1024, D=512, H=1024, 3H=3072, OUT=512.
//
//  - Prologue: weights -> bf16; x + positional encoding -> bf16; zero states.
//  - Per step, per layer: one kernel (16 blocks x 128 thr). Each wave owns a
//    16-wide hidden slice and computes r/z/n gate pre-activations for the
//    whole batch with v_wmma_f32_16x16x32_bf16. Activation panels (64x128)
//    are double-buffered in LDS via global_load_async_to_lds_b128 and shared
//    by all 4 waves; weights stream from global (L2-resident, prefetched).
//  - Epilogue: WMMA output projection.
// ---------------------------------------------------------------------------

typedef __attribute__((ext_vector_type(16))) __bf16 v16bf;
typedef __attribute__((ext_vector_type(8)))  float  v8f;

#define GRU_B    64
#define GRU_D    512
#define GRU_H    1024
#define GRU_S    1024
#define GRU_OUT  512
#define PANEL_K  128   // k-extent of one LDS activation panel (bf16)

static __device__ inline v8f wmma_bf16(v16bf a, v16bf b, v8f c) {
    // 8 args: (neg_a, A, neg_b, B, c_mod, C, reuse_a, reuse_b)
    return __builtin_amdgcn_wmma_f32_16x16x32_bf16(
        false, a, false, b, (short)0, c, false, false);
}

static __device__ inline float sigmoidf_dev(float x) {
    return 1.0f / (1.0f + __expf(-x));
}

// Issue an async copy of a 64-row x 128-col bf16 panel (16 KB) into LDS.
// 1024 x 16B transfers spread over 128 threads (8 per thread).
static __device__ inline void stage_panel_async(const __bf16* __restrict__ src,
                                                int ldk, int kp,
                                                __bf16* dst_lds, int tid) {
    const uint32_t lds_base = (uint32_t)(size_t)dst_lds;
#pragma unroll
    for (int it = 0; it < 8; ++it) {
        const int idx = tid + it * 128;      // 0..1023
        const int row = idx >> 4;            // 0..63
        const int seg = idx & 15;            // 16B segment within row
        const uint64_t g = (uint64_t)(size_t)(src + (size_t)row * ldk + kp)
                           + (uint64_t)(seg * 16);
        const uint32_t l = lds_base + (uint32_t)(row * (PANEL_K * 2) + seg * 16);
        asm volatile("global_load_async_to_lds_b128 %0, %1, off"
                     :: "v"(l), "v"(g) : "memory");
    }
}

static __device__ inline void wait_async_and_sync() {
    asm volatile("s_wait_asynccnt 0x0" ::: "memory");
    __syncthreads();
}

// ---------------------------------------------------------------------------
// f32 -> bf16 conversion (grid-strided)
// ---------------------------------------------------------------------------
__global__ void cvt_f32_bf16_kernel(const float* __restrict__ src,
                                    __bf16* __restrict__ dst, int n) {
    for (int i = blockIdx.x * blockDim.x + threadIdx.x; i < n;
         i += gridDim.x * blockDim.x) {
        dst[i] = (__bf16)src[i];
    }
}

// ---------------------------------------------------------------------------
// x (B,S,D) f32 + positional encoding -> xbf (S,B,D) bf16
// ---------------------------------------------------------------------------
__global__ void prep_x_kernel(const float* __restrict__ x,
                              __bf16* __restrict__ xbf) {
    const int n = GRU_S * GRU_B * GRU_D;
    const float neg_log1e4_over_d = -logf(10000.0f) / (float)GRU_D;
    for (int i = blockIdx.x * blockDim.x + threadIdx.x; i < n;
         i += gridDim.x * blockDim.x) {
        int s = i / (GRU_B * GRU_D);
        int r = i - s * (GRU_B * GRU_D);
        int b = r / GRU_D;
        int d = r - b * GRU_D;
        int i2 = d & ~1;
        float freq = __expf((float)i2 * neg_log1e4_over_d);
        float ang  = (float)s * freq;
        float pe   = (d & 1) ? __cosf(ang) : __sinf(ang);
        float v = x[(size_t)b * GRU_S * GRU_D + (size_t)s * GRU_D + d] + pe;
        xbf[i] = (__bf16)v;
    }
}

// ---------------------------------------------------------------------------
// zero the parity-0 hidden state buffers (f32 + bf16, both layers)
// ---------------------------------------------------------------------------
__global__ void zero_state_kernel(float* __restrict__ h0f, __bf16* __restrict__ h0b,
                                  float* __restrict__ h1f, __bf16* __restrict__ h1b) {
    const int n = GRU_B * GRU_H;
    for (int i = blockIdx.x * blockDim.x + threadIdx.x; i < n;
         i += gridDim.x * blockDim.x) {
        h0f[i] = 0.0f; h1f[i] = 0.0f;
        h0b[i] = (__bf16)0.0f; h1b[i] = (__bf16)0.0f;
    }
}

// ---------------------------------------------------------------------------
// One GRU cell step for one layer.
//   Ain   : (64 x Kin)  bf16 input activations (x_t or h0_new)
//   Hin   : (64 x 1024) bf16 recurrent state (read)
//   Hin_f : (64 x 1024) f32 recurrent state (read, for z*h term)
//   Wih   : (3H x Kin)  bf16,  Whh : (3H x 1024) bf16
//   bih/bhh : (3H) f32
//   Hout_f / Hout_b : (64 x 1024) new state (write)
// Grid: 16 blocks x 128 threads (4 waves); wave owns hidden cols j0..j0+15.
// ---------------------------------------------------------------------------
__global__ __launch_bounds__(128) void gru_step_kernel(
    const __bf16* __restrict__ Ain, int Kin,
    const __bf16* __restrict__ Hin, const float* __restrict__ Hin_f,
    const __bf16* __restrict__ Wih, const __bf16* __restrict__ Whh,
    const float* __restrict__ bih, const float* __restrict__ bhh,
    float* __restrict__ Hout_f, __bf16* __restrict__ Hout_b) {
    __shared__ __align__(16) __bf16 sA[2][GRU_B * PANEL_K];  // 2 x 16 KB

    const int H = GRU_H;
    const int tid  = threadIdx.x;
    const int wave = tid >> 5;
    const int lane = tid & 31;
    const int j0   = (blockIdx.x * 4 + wave) * 16;
    const int jc   = j0 + (lane & 15);       // this lane's hidden column
    const int khalf = (lane >> 4) << 4;      // lanes 16-31 take k+16..k+31
    const int arow  = lane & 15;             // A-fragment row within M-tile

    // bias init (broadcast down each accumulator)
    const float br0 = bih[jc]         + bhh[jc];
    const float bz0 = bih[H + jc]     + bhh[H + jc];
    const float bni = bih[2 * H + jc];
    const float bnh = bhh[2 * H + jc];

    v8f accr[4], accz[4], accni[4], accnh[4];
#pragma unroll
    for (int mt = 0; mt < 4; ++mt) {
#pragma unroll
        for (int e = 0; e < 8; ++e) {
            accr[mt][e]  = br0;
            accz[mt][e]  = bz0;
            accni[mt][e] = bni;
            accnh[mt][e] = bnh;
        }
    }

    // ---- input-side GEMM: gi = Ain @ Wih^T  (K = Kin), LDS-staged A ----
    {
        const __bf16* wr = Wih + (size_t)jc * Kin;
        const __bf16* wz = Wih + (size_t)(H + jc) * Kin;
        const __bf16* wn = Wih + (size_t)(2 * H + jc) * Kin;
        const int npanel = Kin / PANEL_K;

        stage_panel_async(Ain, Kin, 0, sA[0], tid);
        wait_async_and_sync();

        for (int p = 0; p < npanel; ++p) {
            const int buf = p & 1;
            if (p + 1 < npanel)
                stage_panel_async(Ain, Kin, (p + 1) * PANEL_K, sA[buf ^ 1], tid);

            const __bf16* pa = sA[buf];
#pragma unroll
            for (int ks = 0; ks < PANEL_K; ks += 32) {
                const int kk = p * PANEL_K + ks + khalf;  // global k (weights)
                const int lk = ks + khalf;                 // local k (LDS A)
                __builtin_prefetch(wr + kk + 128, 0, 1);
                v16bf a0 = *(const v16bf*)(pa + (arow +  0) * PANEL_K + lk);
                v16bf a1 = *(const v16bf*)(pa + (arow + 16) * PANEL_K + lk);
                v16bf a2 = *(const v16bf*)(pa + (arow + 32) * PANEL_K + lk);
                v16bf a3 = *(const v16bf*)(pa + (arow + 48) * PANEL_K + lk);
                v16bf fr = *(const v16bf*)(wr + kk);
                v16bf fz = *(const v16bf*)(wz + kk);
                v16bf fn = *(const v16bf*)(wn + kk);
                accr[0] = wmma_bf16(a0, fr, accr[0]);
                accr[1] = wmma_bf16(a1, fr, accr[1]);
                accr[2] = wmma_bf16(a2, fr, accr[2]);
                accr[3] = wmma_bf16(a3, fr, accr[3]);
                accz[0] = wmma_bf16(a0, fz, accz[0]);
                accz[1] = wmma_bf16(a1, fz, accz[1]);
                accz[2] = wmma_bf16(a2, fz, accz[2]);
                accz[3] = wmma_bf16(a3, fz, accz[3]);
                accni[0] = wmma_bf16(a0, fn, accni[0]);
                accni[1] = wmma_bf16(a1, fn, accni[1]);
                accni[2] = wmma_bf16(a2, fn, accni[2]);
                accni[3] = wmma_bf16(a3, fn, accni[3]);
            }
            wait_async_and_sync();
        }
    }

    // ---- recurrent-side GEMM: gh = Hin @ Whh^T  (K = 1024), LDS-staged A --
    {
        const __bf16* wr = Whh + (size_t)jc * H;
        const __bf16* wz = Whh + (size_t)(H + jc) * H;
        const __bf16* wn = Whh + (size_t)(2 * H + jc) * H;
        const int npanel = H / PANEL_K;

        stage_panel_async(Hin, H, 0, sA[0], tid);
        wait_async_and_sync();

        for (int p = 0; p < npanel; ++p) {
            const int buf = p & 1;
            if (p + 1 < npanel)
                stage_panel_async(Hin, H, (p + 1) * PANEL_K, sA[buf ^ 1], tid);

            const __bf16* pa = sA[buf];
#pragma unroll
            for (int ks = 0; ks < PANEL_K; ks += 32) {
                const int kk = p * PANEL_K + ks + khalf;
                const int lk = ks + khalf;
                __builtin_prefetch(wr + kk + 128, 0, 1);
                v16bf a0 = *(const v16bf*)(pa + (arow +  0) * PANEL_K + lk);
                v16bf a1 = *(const v16bf*)(pa + (arow + 16) * PANEL_K + lk);
                v16bf a2 = *(const v16bf*)(pa + (arow + 32) * PANEL_K + lk);
                v16bf a3 = *(const v16bf*)(pa + (arow + 48) * PANEL_K + lk);
                v16bf fr = *(const v16bf*)(wr + kk);
                v16bf fz = *(const v16bf*)(wz + kk);
                v16bf fn = *(const v16bf*)(wn + kk);
                accr[0] = wmma_bf16(a0, fr, accr[0]);
                accr[1] = wmma_bf16(a1, fr, accr[1]);
                accr[2] = wmma_bf16(a2, fr, accr[2]);
                accr[3] = wmma_bf16(a3, fr, accr[3]);
                accz[0] = wmma_bf16(a0, fz, accz[0]);
                accz[1] = wmma_bf16(a1, fz, accz[1]);
                accz[2] = wmma_bf16(a2, fz, accz[2]);
                accz[3] = wmma_bf16(a3, fz, accz[3]);
                accnh[0] = wmma_bf16(a0, fn, accnh[0]);
                accnh[1] = wmma_bf16(a1, fn, accnh[1]);
                accnh[2] = wmma_bf16(a2, fn, accnh[2]);
                accnh[3] = wmma_bf16(a3, fn, accnh[3]);
            }
            wait_async_and_sync();
        }
    }

    // ---- GRU nonlinearity + state update (local to wave) ----
    // C/D layout: VGPR e, lanes 0-15 -> M=e, lanes 16-31 -> M=e+8; N=lane&15
    const int mbase = (lane >> 4) << 3;
#pragma unroll
    for (int mt = 0; mt < 4; ++mt) {
#pragma unroll
        for (int e = 0; e < 8; ++e) {
            const int m = mt * 16 + mbase + e;
            const size_t idx = (size_t)m * H + jc;
            float hold = Hin_f[idx];
            float r = sigmoidf_dev(accr[mt][e]);
            float z = sigmoidf_dev(accz[mt][e]);
            float n = tanhf(accni[mt][e] + r * accnh[mt][e]);
            float hn = (1.0f - z) * n + z * hold;
            Hout_f[idx] = hn;
            Hout_b[idx] = (__bf16)hn;
        }
    }
}

// ---------------------------------------------------------------------------
// Output projection: out (64 x 512) = h1 @ w_out^T + b_out, K = 1024
// Grid: 8 blocks x 128 threads; wave owns output cols j0..j0+15.
// ---------------------------------------------------------------------------
__global__ __launch_bounds__(128) void out_proj_kernel(
    const __bf16* __restrict__ Hb, const __bf16* __restrict__ Wo,
    const float* __restrict__ bo, float* __restrict__ out) {
    const int H = GRU_H;
    const int wave = threadIdx.x >> 5;
    const int lane = threadIdx.x & 31;
    const int j0   = (blockIdx.x * 4 + wave) * 16;
    const int jc   = j0 + (lane & 15);
    const int khalf = (lane >> 4) << 4;
    const int arow  = lane & 15;

    const float b0 = bo[jc];
    v8f acc[4];
#pragma unroll
    for (int mt = 0; mt < 4; ++mt)
#pragma unroll
        for (int e = 0; e < 8; ++e) acc[mt][e] = b0;

    const __bf16* w = Wo + (size_t)jc * H;
    for (int k = 0; k < H; k += 32) {
        const int kk = k + khalf;
        v16bf a0 = *(const v16bf*)(Hb + (size_t)(arow +  0) * H + kk);
        v16bf a1 = *(const v16bf*)(Hb + (size_t)(arow + 16) * H + kk);
        v16bf a2 = *(const v16bf*)(Hb + (size_t)(arow + 32) * H + kk);
        v16bf a3 = *(const v16bf*)(Hb + (size_t)(arow + 48) * H + kk);
        v16bf f  = *(const v16bf*)(w + kk);
        acc[0] = wmma_bf16(a0, f, acc[0]);
        acc[1] = wmma_bf16(a1, f, acc[1]);
        acc[2] = wmma_bf16(a2, f, acc[2]);
        acc[3] = wmma_bf16(a3, f, acc[3]);
    }

    const int mbase = (lane >> 4) << 3;
#pragma unroll
    for (int mt = 0; mt < 4; ++mt)
#pragma unroll
        for (int e = 0; e < 8; ++e) {
            const int m = mt * 16 + mbase + e;
            out[(size_t)m * GRU_OUT + jc] = acc[mt][e];
        }
}

// ---------------------------------------------------------------------------
// host launcher
// ---------------------------------------------------------------------------
extern "C" void kernel_launch(void* const* d_in, const int* in_sizes, int n_in,
                              void* d_out, int out_size, void* d_ws, size_t ws_size,
                              hipStream_t stream) {
    (void)in_sizes; (void)n_in; (void)out_size; (void)ws_size;

    const float* x     = (const float*)d_in[0];
    const float* w_ih0 = (const float*)d_in[1];
    const float* w_hh0 = (const float*)d_in[2];
    const float* b_ih0 = (const float*)d_in[3];
    const float* b_hh0 = (const float*)d_in[4];
    const float* w_ih1 = (const float*)d_in[5];
    const float* w_hh1 = (const float*)d_in[6];
    const float* b_ih1 = (const float*)d_in[7];
    const float* b_hh1 = (const float*)d_in[8];
    const float* w_out = (const float*)d_in[9];
    const float* b_out = (const float*)d_in[10];
    float* out = (float*)d_out;

    // ---- workspace layout (all offsets comfortably aligned) ----
    char* ws = (char*)d_ws;
    size_t o = 0;
    __bf16* xbf = (__bf16*)(ws + o); o += (size_t)GRU_S * GRU_B * GRU_D * 2;   // 64 MiB
    __bf16* wih0b = (__bf16*)(ws + o); o += (size_t)3 * GRU_H * GRU_D * 2;     // 3 MiB
    __bf16* whh0b = (__bf16*)(ws + o); o += (size_t)3 * GRU_H * GRU_H * 2;     // 6 MiB
    __bf16* wih1b = (__bf16*)(ws + o); o += (size_t)3 * GRU_H * GRU_H * 2;     // 6 MiB
    __bf16* whh1b = (__bf16*)(ws + o); o += (size_t)3 * GRU_H * GRU_H * 2;     // 6 MiB
    __bf16* woutb = (__bf16*)(ws + o); o += (size_t)GRU_OUT * GRU_H * 2;       // 1 MiB
    float*  h0f[2]; __bf16* h0b[2];
    float*  h1f[2]; __bf16* h1b[2];
    for (int p = 0; p < 2; ++p) { h0f[p] = (float*)(ws + o);  o += (size_t)GRU_B * GRU_H * 4; }
    for (int p = 0; p < 2; ++p) { h0b[p] = (__bf16*)(ws + o); o += (size_t)GRU_B * GRU_H * 2; }
    for (int p = 0; p < 2; ++p) { h1f[p] = (float*)(ws + o);  o += (size_t)GRU_B * GRU_H * 4; }
    for (int p = 0; p < 2; ++p) { h1b[p] = (__bf16*)(ws + o); o += (size_t)GRU_B * GRU_H * 2; }

    // ---- prologue: weight conversion, x prep, state zero ----
    {
        int n;
        n = 3 * GRU_H * GRU_D;
        cvt_f32_bf16_kernel<<<(n + 255) / 256, 256, 0, stream>>>(w_ih0, wih0b, n);
        n = 3 * GRU_H * GRU_H;
        cvt_f32_bf16_kernel<<<(n + 255) / 256, 256, 0, stream>>>(w_hh0, whh0b, n);
        cvt_f32_bf16_kernel<<<(n + 255) / 256, 256, 0, stream>>>(w_hh1 ? w_ih1 : w_ih1, wih1b, n);
        cvt_f32_bf16_kernel<<<(n + 255) / 256, 256, 0, stream>>>(w_hh1, whh1b, n);
        n = GRU_OUT * GRU_H;
        cvt_f32_bf16_kernel<<<(n + 255) / 256, 256, 0, stream>>>(w_out, woutb, n);

        int nx = GRU_S * GRU_B * GRU_D;
        prep_x_kernel<<<(nx + 255) / 256, 256, 0, stream>>>(x, xbf);

        int nh = GRU_B * GRU_H;
        zero_state_kernel<<<(nh + 255) / 256, 256, 0, stream>>>(
            h0f[0], h0b[0], h1f[0], h1b[0]);
    }

    // ---- sequential recurrence: 1024 steps x 2 layers ----
    for (int s = 0; s < GRU_S; ++s) {
        const int p = s & 1;          // read parity
        const int q = 1 - p;          // write parity
        const __bf16* xs = xbf + (size_t)s * GRU_B * GRU_D;
        // layer 0: input = x_t (K=512)
        gru_step_kernel<<<16, 128, 0, stream>>>(
            xs, GRU_D, h0b[p], h0f[p], wih0b, whh0b, b_ih0, b_hh0,
            h0f[q], h0b[q]);
        // layer 1: input = h0_new (K=1024)
        gru_step_kernel<<<16, 128, 0, stream>>>(
            h0b[q], GRU_H, h1b[p], h1f[p], wih1b, whh1b, b_ih1, b_hh1,
            h1f[q], h1b[q]);
    }

    // step 1023 wrote parity 0 -> final h1 is h1b[0]
    out_proj_kernel<<<8, 128, 0, stream>>>(h1b[0], woutb, b_out, out);
}
